// PointTokenizer_16716012716431
// MI455X (gfx1250) — matrix operand order repaired and checked by reading, not compiled, not run on hardware
//
#include <hip/hip_runtime.h>
#include <hip/hip_bf16.h>

typedef __attribute__((ext_vector_type(16))) _Float16 v16h;
typedef __attribute__((ext_vector_type(8)))  _Float16 v8h;
typedef __attribute__((ext_vector_type(8)))  float    v8f;

#define NB   16          // batches
#define NP   16384       // points per batch
#define NC   256         // n_centers (M)
#define KNN  32          // k
#define EMB  256
#define TOKN (NB * NC * EMB)   // tokens element count in d_out
#define BNEPS 1e-5f

// ---------------------------------------------------------------------------
// Kernel 1: furthest point sampling. One block per batch. Running min-distance
// array lives in LDS; argmax via tree reduction with first-index tie-break
// (matches jnp.argmax). Centers written directly to d_out tail.
// ---------------------------------------------------------------------------
__global__ void fps_kernel(const float* __restrict__ xyz,
                           float* __restrict__ out_centers) {
  extern __shared__ char smem[];
  float* dist = (float*)smem;                       // NP floats (64 KB)
  float* rv   = (float*)(smem + NP * sizeof(float));
  int*   ri   = (int*)(rv + 256);

  const int b = blockIdx.x;
  const int t = threadIdx.x;
  const float* xb = xyz + (size_t)b * NP * 3;

  for (int i = t; i < NP; i += 256) dist[i] = __builtin_inff();
  __syncthreads();

  int far = 0;
  for (int it = 0; it < NC; ++it) {
    // record current farthest as center #it (uniform scalar loads)
    const float cx = xb[far * 3 + 0];
    const float cy = xb[far * 3 + 1];
    const float cz = xb[far * 3 + 2];
    if (t == 0) {
      float* oc = out_centers + ((size_t)b * NC + it) * 3;
      oc[0] = cx; oc[1] = cy; oc[2] = cz;
    }
    float lm = -1.0f; int li = 0;
    for (int i = t; i < NP; i += 256) {
      const float dx = xb[i * 3 + 0] - cx;
      const float dy = xb[i * 3 + 1] - cy;
      const float dz = xb[i * 3 + 2] - cz;
      const float d  = dx * dx + dy * dy + dz * dz;
      float nd = dist[i];
      nd = (d < nd) ? d : nd;
      dist[i] = nd;
      if (nd > lm) { lm = nd; li = i; }   // strict >: keeps lowest index
    }
    rv[t] = lm; ri[t] = li;
    __syncthreads();
    for (int s = 128; s > 0; s >>= 1) {
      if (t < s) {
        const float ov = rv[t + s]; const int oi = ri[t + s];
        if (ov > rv[t] || (ov == rv[t] && oi < ri[t])) { rv[t] = ov; ri[t] = oi; }
      }
      __syncthreads();
    }
    far = ri[0];
    __syncthreads();
  }
}

// ---------------------------------------------------------------------------
// Kernel 2: kNN grouping. One block per (batch, center). All NP distances in
// LDS, then 32 iterative argmin reductions (first-index tie-break matches
// lax.top_k). Emits center-relative patch coords. Order is irrelevant since
// downstream is a max-pool over k.
// ---------------------------------------------------------------------------
__global__ void knn_kernel(const float* __restrict__ xyz,
                           const float* __restrict__ centers,
                           float* __restrict__ patches) {
  extern __shared__ char smem[];
  float* dist = (float*)smem;
  float* rv   = (float*)(smem + NP * sizeof(float));
  int*   ri   = (int*)(rv + 256);

  const int bm = blockIdx.x;          // b * NC + m
  const int b  = bm >> 8;             // NC == 256
  const int t  = threadIdx.x;
  const float* xb = xyz + (size_t)b * NP * 3;
  const float cx = centers[bm * 3 + 0];
  const float cy = centers[bm * 3 + 1];
  const float cz = centers[bm * 3 + 2];

  for (int i = t; i < NP; i += 256) {
    const float dx = xb[i * 3 + 0] - cx;
    const float dy = xb[i * 3 + 1] - cy;
    const float dz = xb[i * 3 + 2] - cz;
    dist[i] = dx * dx + dy * dy + dz * dz;
  }
  __syncthreads();

  for (int j = 0; j < KNN; ++j) {
    float lm = __builtin_inff(); int li = 0;
    for (int i = t; i < NP; i += 256) {
      const float d = dist[i];
      if (d < lm) { lm = d; li = i; }
    }
    rv[t] = lm; ri[t] = li;
    __syncthreads();
    for (int s = 128; s > 0; s >>= 1) {
      if (t < s) {
        const float ov = rv[t + s]; const int oi = ri[t + s];
        if (ov < rv[t] || (ov == rv[t] && oi < ri[t])) { rv[t] = ov; ri[t] = oi; }
      }
      __syncthreads();
    }
    const int sel = ri[0];
    if (t == 0) {
      float* pp = patches + ((size_t)bm * KNN + j) * 3;
      pp[0] = xb[sel * 3 + 0] - cx;
      pp[1] = xb[sel * 3 + 1] - cy;
      pp[2] = xb[sel * 3 + 2] - cz;
      dist[sel] = __builtin_inff();
    }
    __syncthreads();
  }
}

// ---------------------------------------------------------------------------
// Kernel 3a: zero the token area so the ReLU>=0 atomic-max pool is correct.
// ---------------------------------------------------------------------------
__global__ void zero_kernel(float* __restrict__ p, int n) {
  for (int i = blockIdx.x * blockDim.x + threadIdx.x; i < n;
       i += gridDim.x * blockDim.x)
    p[i] = 0.0f;
}

// ---------------------------------------------------------------------------
// Kernel 3b: mini-PointNet via v_wmma_f32_16x16x32_f16.
// 256 threads = 8 waves; each wave owns 16 points (one 16-row M tile).
// Weights staged transposed (Wt[n][k], f16) in LDS; activations restaged
// per-wave in LDS between layers in A-fragment-friendly row-major f16.
// Max-pool over k=32 done with integer atomicMax on non-negative f32 bits.
// ---------------------------------------------------------------------------
__device__ __forceinline__ v16h make_afrag(const _Float16* row, int khi) {
  // Contiguous K layout: lanes 0-15 -> K 0..15,
  // lanes 16-31 -> K 16..31 (two contiguous 16B LDS loads).
  v8h lo = *(const v8h*)(row + khi * 16);
  v8h hi = *(const v8h*)(row + khi * 16 + 8);
  v16h a;
#pragma unroll
  for (int i = 0; i < 8; ++i) { a[i] = lo[i]; a[8 + i] = hi[i]; }
  return a;
}

__device__ __forceinline__ v16h make_bfrag(const _Float16* wrow) {
  // b[j] = W[koff + j][n], 16 contiguous halves of the transposed weight row.
  v8h lo = *(const v8h*)(wrow);
  v8h hi = *(const v8h*)(wrow + 8);
  v16h b;
#pragma unroll
  for (int i = 0; i < 8; ++i) { b[i] = lo[i]; b[8 + i] = hi[i]; }
  return b;
}

__global__ void mlp_kernel(const float* __restrict__ patches,
                           const float* __restrict__ W1, const float* __restrict__ b1,
                           const float* __restrict__ g1, const float* __restrict__ bt1,
                           const float* __restrict__ W2, const float* __restrict__ b2,
                           const float* __restrict__ g2, const float* __restrict__ bt2,
                           const float* __restrict__ W3, const float* __restrict__ b3,
                           const float* __restrict__ g3, const float* __restrict__ bt3,
                           float* __restrict__ tokens) {
  extern __shared__ char smem[];
  _Float16* Wt1  = (_Float16*)smem;        // 64  x 32  (K padded 3->32, zero fill)
  _Float16* Wt2  = Wt1 + 64 * 32;          // 128 x 64
  _Float16* Wt3  = Wt2 + 128 * 64;         // 256 x 128
  _Float16* actA = Wt3 + 256 * 128;        // 8 waves x 16 x 64
  _Float16* actB = actA + 8 * 16 * 64;     // 8 waves x 16 x 128

  const int t = threadIdx.x;

  // Stage weights transposed into LDS as f16
  for (int i = t; i < 64 * 32; i += 256) {
    const int n = i >> 5, kk = i & 31;
    Wt1[i] = (kk < 3) ? (_Float16)W1[kk * 64 + n] : (_Float16)0.0f;
  }
  for (int i = t; i < 128 * 64; i += 256) {
    const int n = i >> 6, kk = i & 63;
    Wt2[i] = (_Float16)W2[kk * 128 + n];
  }
  for (int i = t; i < 256 * 128; i += 256) {
    const int n = i >> 7, kk = i & 127;
    Wt3[i] = (_Float16)W3[kk * 256 + n];
  }
  __syncthreads();

  const int lane = t & 31;
  const int wv   = t >> 5;
  const int mrow = lane & 15;
  const int khi  = lane >> 4;    // 0|1 -> which K half-group this lane holds
  const int ncol = lane & 15;    // D-matrix column for this lane
  const int pbase = blockIdx.x * 128 + wv * 16;
  _Float16* aA = actA + wv * (16 * 64);
  _Float16* aB = actB + wv * (16 * 128);
  const float inv = rsqrtf(1.0f + BNEPS);

  // ---- Layer 1: 3 -> 64, single k-step (K padded to 32, zeros elsewhere) ----
  v16h a1 = {};
  if (khi == 0) {   // only lanes holding K=0..7 carry the xyz features
    const float* pp = patches + (size_t)(pbase + mrow) * 3;
    a1[0] = (_Float16)pp[0];
    a1[1] = (_Float16)pp[1];
    a1[2] = (_Float16)pp[2];
  }
#pragma unroll
  for (int nt = 0; nt < 4; ++nt) {
    v16h bf = make_bfrag(Wt1 + (nt * 16 + ncol) * 32 + khi * 16);
    v8f c = {};
    c = __builtin_amdgcn_wmma_f32_16x16x32_f16(false, a1, false, bf,
                                               (short)0, c, false, false);
    const int n = nt * 16 + ncol;
    const float s = g1[n] * inv, bb = b1[n], bt = bt1[n];
#pragma unroll
    for (int r = 0; r < 8; ++r) {
      float y = (c[r] + bb) * s + bt;
      y = y > 0.0f ? y : 0.0f;
      aA[(r + khi * 8) * 64 + n] = (_Float16)y;
    }
  }
  asm volatile("s_wait_dscnt 0x0" ::: "memory");  // intra-wave LDS RAW

  // ---- Layer 2: 64 -> 128, two k-steps ----
#pragma unroll
  for (int nt = 0; nt < 8; ++nt) {
    v8f c = {};
#pragma unroll
    for (int ks = 0; ks < 2; ++ks) {
      v16h a  = make_afrag(aA + mrow * 64 + ks * 32, khi);
      v16h bf = make_bfrag(Wt2 + (nt * 16 + ncol) * 64 + ks * 32 + khi * 16);
      c = __builtin_amdgcn_wmma_f32_16x16x32_f16(false, a, false, bf,
                                                 (short)0, c, false, false);
    }
    const int n = nt * 16 + ncol;
    const float s = g2[n] * inv, bb = b2[n], bt = bt2[n];
#pragma unroll
    for (int r = 0; r < 8; ++r) {
      float y = (c[r] + bb) * s + bt;
      y = y > 0.0f ? y : 0.0f;
      aB[(r + khi * 8) * 128 + n] = (_Float16)y;
    }
  }
  asm volatile("s_wait_dscnt 0x0" ::: "memory");

  // ---- Layer 3: 128 -> 256, four k-steps; fused max-pool over k=32 ----
#pragma unroll
  for (int nt = 0; nt < 16; ++nt) {
    v8f c = {};
#pragma unroll
    for (int ks = 0; ks < 4; ++ks) {
      v16h a  = make_afrag(aB + mrow * 128 + ks * 32, khi);
      v16h bf = make_bfrag(Wt3 + (nt * 16 + ncol) * 128 + ks * 32 + khi * 16);
      c = __builtin_amdgcn_wmma_f32_16x16x32_f16(false, a, false, bf,
                                                 (short)0, c, false, false);
    }
    const int n = nt * 16 + ncol;
    const float s = g3[n] * inv, bb = b3[n], bt = bt3[n];
#pragma unroll
    for (int r = 0; r < 8; ++r) {
      float y = (c[r] + bb) * s + bt;
      y = y > 0.0f ? y : 0.0f;                       // >= 0 -> uint order == float order
      const int p = pbase + r + khi * 8;             // global point index
      unsigned int* dst =
          (unsigned int*)tokens + (size_t)(p >> 5) * EMB + n;  // p/KNN
      atomicMax(dst, __float_as_uint(y));
    }
  }
}

extern "C" void kernel_launch(void* const* d_in, const int* in_sizes, int n_in,
                              void* d_out, int out_size, void* d_ws, size_t ws_size,
                              hipStream_t stream) {
  const float* xyz = (const float*)d_in[0];
  const float* W1  = (const float*)d_in[1];
  const float* b1  = (const float*)d_in[2];
  const float* g1  = (const float*)d_in[3];
  const float* bt1 = (const float*)d_in[4];
  const float* W2  = (const float*)d_in[5];
  const float* b2  = (const float*)d_in[6];
  const float* g2  = (const float*)d_in[7];
  const float* bt2 = (const float*)d_in[8];
  const float* W3  = (const float*)d_in[9];
  const float* b3  = (const float*)d_in[10];
  const float* g3  = (const float*)d_in[11];
  const float* bt3 = (const float*)d_in[12];

  float* out     = (float*)d_out;
  float* out_ctr = out + TOKN;                 // centers follow tokens
  float* patches = (float*)d_ws;               // (B*M*k*3) f32 scratch

  const size_t red_smem = (size_t)NP * 4 + 256 * 4 + 256 * 4;   // 67584 B
  const size_t mlp_smem =
      (size_t)(64 * 32 + 128 * 64 + 256 * 128 + 8 * 16 * 64 + 8 * 16 * 128) * 2; // 135168 B

  fps_kernel<<<NB, 256, red_smem, stream>>>(xyz, out_ctr);
  knn_kernel<<<NB * NC, 256, red_smem, stream>>>(xyz, out_ctr, patches);
  zero_kernel<<<512, 256, 0, stream>>>(out, TOKN);
  mlp_kernel<<<(NB * NC * KNN) / 128, 256, mlp_smem, stream>>>(
      patches, W1, b1, g1, bt1, W2, b2, g2, bt2, W3, b3, g3, bt3, out);
}
